// CodeGEMMLinear_59785944760454
// MI455X (gfx1250) — matrix operand order repaired and checked by reading, not compiled, not run on hardware
//
#include <hip/hip_runtime.h>
#include <hip/hip_bf16.h>

// ---------------------------------------------------------------------------
// AQLM-style quantized linear for MI455X (gfx1250, wave32, WMMA).
// Phase 1: dequantize weight -> f16 W^T (O-major) in workspace (32 MB)
// Phase 2: convert x f32 -> f16 (16 MB)
// Phase 3: WMMA GEMM, 32x64 tile per wave, v_wmma_f32_16x16x32_f16,
//          whole 48 MB working set is L2-resident (192 MB L2).
// ---------------------------------------------------------------------------

typedef __attribute__((ext_vector_type(16))) _Float16 v16h;
typedef __attribute__((ext_vector_type(8)))  _Float16 v8h;
typedef __attribute__((ext_vector_type(4)))  _Float16 v4h;
typedef __attribute__((ext_vector_type(8)))  float    v8f;

#define NUM_CB   2
#define IN_F     4096
#define OUT_F    4096
#define VEC_LEN  8
#define CB_SIZE  256
#define M_ROWS   2048          // B*T
#define P4       (IN_F / VEC_LEN / 4)   // 128 packed int32 per column per codebook

// ---------------------------------------------------------------------------
// Kernel 1: dequant.  One thread per (o, p4): unpack 2x4 byte codes,
// gather+sum codebook vectors, scale, emit 32 contiguous f16 of W^T[o][k].
// Thread mapping puts p4 in the low bits so 128 consecutive threads write one
// fully-contiguous 8 KB row segment (coalesced stores).
// ---------------------------------------------------------------------------
__global__ __launch_bounds__(256) void aqlm_dequant_kernel(
    const int* __restrict__ codes,       // (2, 128, 4096)
    const float* __restrict__ codebooks, // (2, 256, 8)
    const float* __restrict__ scales,    // (32, 4096)
    _Float16* __restrict__ wt)           // (4096 o, 4096 k) f16
{
  const int tid = blockIdx.x * blockDim.x + threadIdx.x;  // 524288 total
  const int o   = tid >> 7;     // 0..4095
  const int p4  = tid & 127;    // 0..127

  const int c0 = codes[p4 * OUT_F + o];
  const int c1 = codes[P4 * OUT_F + p4 * OUT_F + o];
  // 32 k-values of one p4 always share one group of 128 -> single scale load.
  const float s = scales[(p4 >> 2) * OUT_F + o];

  _Float16* __restrict__ dst = wt + (size_t)o * IN_F + p4 * 32;

#pragma unroll
  for (int j = 0; j < 4; ++j) {
    const int i0 = (c0 >> (8 * j)) & 0xFF;
    const int i1 = (c1 >> (8 * j)) & 0xFF;
    const float4* e0 = (const float4*)(codebooks + i0 * VEC_LEN);
    const float4* e1 = (const float4*)(codebooks + CB_SIZE * VEC_LEN + i1 * VEC_LEN);
    const float4 a0 = e0[0], a1 = e0[1];
    const float4 b0 = e1[0], b1 = e1[1];
    v8h h;
    h[0] = (_Float16)((a0.x + b0.x) * s);
    h[1] = (_Float16)((a0.y + b0.y) * s);
    h[2] = (_Float16)((a0.z + b0.z) * s);
    h[3] = (_Float16)((a0.w + b0.w) * s);
    h[4] = (_Float16)((a1.x + b1.x) * s);
    h[5] = (_Float16)((a1.y + b1.y) * s);
    h[6] = (_Float16)((a1.z + b1.z) * s);
    h[7] = (_Float16)((a1.w + b1.w) * s);
    *(v8h*)(dst + j * 8) = h;
  }
}

// ---------------------------------------------------------------------------
// Kernel 2: x f32 -> f16, 4 elements per thread.
// ---------------------------------------------------------------------------
__global__ __launch_bounds__(256) void aqlm_cvt_x_kernel(
    const float* __restrict__ x, _Float16* __restrict__ x16)
{
  const int i = blockIdx.x * blockDim.x + threadIdx.x;  // n/4 threads
  const float4 v = ((const float4*)x)[i];
  v4h h;
  h[0] = (_Float16)v.x; h[1] = (_Float16)v.y;
  h[2] = (_Float16)v.z; h[3] = (_Float16)v.w;
  ((v4h*)x16)[i] = h;
}

// ---------------------------------------------------------------------------
// Kernel 3: WMMA GEMM  y[m][n] = sum_k A[m][k] * Wt[n][k]
// Per wave: 32 (M) x 64 (N) tile -> 2 A frags, 4 B frags, 8 accumulators.
// Fragment addressing (this code's contiguous K order, same for A and B):
//   A 16x32 f16 : lane<16 holds M=lane,  K = k0+[0..15] contiguous
//                 lane>=16 holds M=lane-16, K = k0+[16..31] contiguous
//   B 32x16 f16 : lane<16 holds N=lane,  K = k0+[0..15] contiguous
//                 lane>=16 holds N=lane-16, K = k0+[16..31] contiguous
//   C/D 16x16 f32: VGPR r = row r (lanes 0-15) / row r+8 (lanes 16-31)
// ---------------------------------------------------------------------------
__device__ __forceinline__ v16h load_a_frag(const _Float16* p) {
  const v8h lo = *(const v8h*)(p);        // K = base..base+7
  const v8h hi = *(const v8h*)(p + 8);    // K = base+8..base+15
  return __builtin_shufflevector(lo, hi, 0, 1, 2, 3, 4, 5, 6, 7,
                                 8, 9, 10, 11, 12, 13, 14, 15);
}
__device__ __forceinline__ v16h load_b_frag(const _Float16* p) {
  const v8h lo = *(const v8h*)(p);        // K contiguous
  const v8h hi = *(const v8h*)(p + 8);
  return __builtin_shufflevector(lo, hi, 0, 1, 2, 3, 4, 5, 6, 7,
                                 8, 9, 10, 11, 12, 13, 14, 15);
}

__global__ __launch_bounds__(256) void aqlm_wmma_gemm_kernel(
    const _Float16* __restrict__ A,   // (2048, 4096) f16 row-major
    const _Float16* __restrict__ Bt,  // (4096 o, 4096 k) f16
    float* __restrict__ C)            // (2048, 4096) f32
{
  const int lane = threadIdx.x & 31;
  const int wave = blockIdx.x * (blockDim.x >> 5) + (threadIdx.x >> 5);
  const int nTilesN = OUT_F / 64;            // 64
  const int m0 = (wave / nTilesN) * 32;
  const int n0 = (wave % nTilesN) * 64;

  const int lh = lane & 15;                  // 0..15
  const int hs = lane >> 4;                  // 0 or 1

  const _Float16* aP0 = A + (size_t)(m0 + lh) * IN_F + hs * 16;
  const _Float16* aP1 = aP0 + (size_t)16 * IN_F;
  const _Float16* bP0 = Bt + (size_t)(n0 +  0 + lh) * IN_F + hs * 16;
  const _Float16* bP1 = Bt + (size_t)(n0 + 16 + lh) * IN_F + hs * 16;
  const _Float16* bP2 = Bt + (size_t)(n0 + 32 + lh) * IN_F + hs * 16;
  const _Float16* bP3 = Bt + (size_t)(n0 + 48 + lh) * IN_F + hs * 16;

  v8f acc00 = {}, acc01 = {}, acc02 = {}, acc03 = {};
  v8f acc10 = {}, acc11 = {}, acc12 = {}, acc13 = {};

  for (int k0 = 0; k0 < IN_F; k0 += 32) {
    // hint the next K slice into cache (global_prefetch_b8)
    __builtin_prefetch(aP0 + k0 + 32, 0, 3);
    __builtin_prefetch(bP0 + k0 + 32, 0, 3);
    __builtin_prefetch(bP2 + k0 + 32, 0, 3);

    const v16h a0 = load_a_frag(aP0 + k0);
    const v16h a1 = load_a_frag(aP1 + k0);

    v16h b;
    b = load_b_frag(bP0 + k0);
    acc00 = __builtin_amdgcn_wmma_f32_16x16x32_f16(false, a0, false, b,
                                                   (short)0, acc00, false, false);
    acc10 = __builtin_amdgcn_wmma_f32_16x16x32_f16(false, a1, false, b,
                                                   (short)0, acc10, false, false);
    b = load_b_frag(bP1 + k0);
    acc01 = __builtin_amdgcn_wmma_f32_16x16x32_f16(false, a0, false, b,
                                                   (short)0, acc01, false, false);
    acc11 = __builtin_amdgcn_wmma_f32_16x16x32_f16(false, a1, false, b,
                                                   (short)0, acc11, false, false);
    b = load_b_frag(bP2 + k0);
    acc02 = __builtin_amdgcn_wmma_f32_16x16x32_f16(false, a0, false, b,
                                                   (short)0, acc02, false, false);
    acc12 = __builtin_amdgcn_wmma_f32_16x16x32_f16(false, a1, false, b,
                                                   (short)0, acc12, false, false);
    b = load_b_frag(bP3 + k0);
    acc03 = __builtin_amdgcn_wmma_f32_16x16x32_f16(false, a0, false, b,
                                                   (short)0, acc03, false, false);
    acc13 = __builtin_amdgcn_wmma_f32_16x16x32_f16(false, a1, false, b,
                                                   (short)0, acc13, false, false);
  }

  // Store: VGPR r of a 16x16 f32 C/D tile holds row (r + 8*hs), col = lh.
#pragma unroll
  for (int r = 0; r < 8; ++r) {
    const size_t row0 = (size_t)(m0 + hs * 8 + r) * OUT_F;        // A-frag 0
    const size_t row1 = (size_t)(m0 + 16 + hs * 8 + r) * OUT_F;   // A-frag 1
    C[row0 + n0 +  0 + lh] = acc00[r];
    C[row0 + n0 + 16 + lh] = acc01[r];
    C[row0 + n0 + 32 + lh] = acc02[r];
    C[row0 + n0 + 48 + lh] = acc03[r];
    C[row1 + n0 +  0 + lh] = acc10[r];
    C[row1 + n0 + 16 + lh] = acc11[r];
    C[row1 + n0 + 32 + lh] = acc12[r];
    C[row1 + n0 + 48 + lh] = acc13[r];
  }
}

// ---------------------------------------------------------------------------
extern "C" void kernel_launch(void* const* d_in, const int* in_sizes, int n_in,
                              void* d_out, int out_size, void* d_ws, size_t ws_size,
                              hipStream_t stream) {
  const float* x         = (const float*)d_in[0];  // (1, 2048, 4096) f32
  const int*   codes     = (const int*)d_in[1];    // (2, 128, 4096) i32
  const float* codebooks = (const float*)d_in[2];  // (2, 256, 8) f32
  const float* scales    = (const float*)d_in[3];  // (32, 4096) f32
  float*       y         = (float*)d_out;          // (1, 2048, 4096) f32

  _Float16* wt  = (_Float16*)d_ws;                                    // 32 MB
  _Float16* x16 = (_Float16*)((char*)d_ws + (size_t)OUT_F * IN_F * 2); // 16 MB

  // Phase 1: dequantize weight -> f16 W^T (O-major).
  {
    const int threads = OUT_F * P4;                 // 524288
    aqlm_dequant_kernel<<<threads / 256, 256, 0, stream>>>(codes, codebooks,
                                                           scales, wt);
  }
  // Phase 2: x -> f16.
  {
    const int n4 = (M_ROWS * IN_F) / 4;             // 2097152
    aqlm_cvt_x_kernel<<<n4 / 256, 256, 0, stream>>>(x, x16);
  }
  // Phase 3: WMMA GEMM. 32 (M) x 64 (N) wave-tiles = 4096 waves, 8 per block.
  {
    const int waves = (M_ROWS / 32) * (OUT_F / 64); // 4096
    aqlm_wmma_gemm_kernel<<<waves / 8, 256, 0, stream>>>(x16, wt, y);
  }
}
